// OrderedMemoryDecoder_40510131536274
// MI455X (gfx1250) — matrix operand order repaired and checked by reading, not compiled, not run on hardware
//
#include <hip/hip_runtime.h>
#include <vector>
#include <cmath>
#include <cstdint>
#include <cstring>
#include <utility>

// ---------------------------------------------------------------------------
// Problem constants (match reference: B=8, H=1024, V=32000, DEPTH=14)
// ---------------------------------------------------------------------------
#define BB    8
#define HD    1024
#define VV    32000
#define DEPTH 14
#define LN_EPS 1e-5f

typedef __attribute__((ext_vector_type(16))) __bf16 v16bf;
typedef __attribute__((ext_vector_type(8)))  float  v8f;

// ---------------------------------------------------------------------------
// Host: replicate _structure(depth) exactly (static, computed once at load)
// ---------------------------------------------------------------------------
namespace omd {

struct RecOut {
    std::vector<std::vector<int>> masks;           // per level, len 2K entries (0/1)
    std::vector<std::vector<int>> Bl, Br;
    std::vector<std::pair<int,int>> S;
};

static RecOut rec(int idx_start, int n, const std::vector<int>& lengths) {
    int K = (int)lengths.size();
    RecOut out;
    out.Bl.resize(K); out.Br.resize(K);
    std::vector<int> idxs(K);
    for (int i = 0; i < K; ++i) {
        idxs[i] = idx_start + i;
        out.Bl[i] = { idxs[i] };
        out.Br[i] = { idxs[i] };
    }
    if (n > 1) {
        std::vector<int> pmask(K), mask;
        std::vector<int> child_lengths;
        for (int i = 0; i < K; ++i) {
            int L = lengths[i];
            int m = (L - 1) > 0 ? 1 : 0;
            pmask[i] = m;
            mask.push_back(m); mask.push_back(m);
            if (m) {
                child_lengths.push_back((int)std::ceil(std::pow((double)(L - 1), 0.75)));
                child_lengths.push_back(L - 1);
            }
        }
        RecOut below = rec(idxs[K - 1] + 1, n - 1, child_lengths);
        out.masks.push_back(mask);
        for (auto& mm : below.masks) out.masks.push_back(mm);
        out.S = below.S;
        int j = 0;
        for (int i = 0; i < K; ++i) {
            if (pmask[i]) {
                for (int v : below.Bl[j])     out.Bl[i].push_back(v);
                for (int v : below.Br[j + 1]) out.Br[i].push_back(v);
                for (int vi : below.Br[j])
                    for (int vj : below.Bl[j + 1])
                        out.S.push_back({vi, vj});
                j += 2;
            }
        }
    }
    return out;
}

struct Structure {
    int T;                                  // total nodes (351 for depth 14)
    std::vector<int> levelK;                // nodes per level        (DEPTH entries)
    std::vector<int> levelOff;              // row offset per level   (DEPTH entries)
    std::vector<int> childP, childS;        // per child (levels 1..): parent local idx, side
    std::vector<int> childBase;             // offset into childP per level transition
    std::vector<int> Bl0, Br0;              // start_idxs / end_idxs
    std::vector<int> Sout, Sin;             // out_idxs / in_idxs
};

static Structure buildStructure() {
    Structure st;
    RecOut r = rec(0, DEPTH, std::vector<int>{DEPTH});
    st.levelK.push_back(1);
    st.levelOff.push_back(0);
    for (size_t l = 0; l < r.masks.size(); ++l) {
        const auto& mask = r.masks[l];
        st.childBase.push_back((int)st.childP.size());
        int kc = 0;
        for (size_t i = 0; i < mask.size(); ++i) {
            if (mask[i]) {
                st.childP.push_back((int)(i / 2));
                st.childS.push_back((int)(i % 2));
                ++kc;
            }
        }
        st.levelOff.push_back(st.levelOff.back() + st.levelK.back());
        st.levelK.push_back(kc);
    }
    st.T = st.levelOff.back() + st.levelK.back();
    st.Bl0 = r.Bl[0];
    st.Br0 = r.Br[0];
    for (auto& p : r.S) { st.Sout.push_back(p.first); st.Sin.push_back(p.second); }
    return st;
}

static const Structure g_S = buildStructure();

static size_t alignUp(size_t v, size_t a) { return (v + a - 1) & ~(a - 1); }

} // namespace omd

// ---------------------------------------------------------------------------
// Device helpers
// ---------------------------------------------------------------------------
__device__ __forceinline__ __bf16 f2bf(float f) {
    union { float f; unsigned u; } in; in.f = f;
    unsigned u = in.u;
    if ((u & 0x7F800000u) != 0x7F800000u)        // not inf/nan: round-to-nearest-even
        u += 0x7FFFu + ((u >> 16) & 1u);
    union { unsigned short s; __bf16 b; } o;
    o.s = (unsigned short)(u >> 16);
    return o.b;
}

// ---------------------------------------------------------------------------
// Kernel: out[b][j] = sum_k x[b][k] * W[k][j] + bias[j]      (grid = B rows)
// Used for h0 = x@W_in + b_in  and  g = x@W_g + b_g
// ---------------------------------------------------------------------------
__global__ __launch_bounds__(1024)
void k_in(const float* __restrict__ x, const float* __restrict__ W,
          const float* __restrict__ bias, float* __restrict__ out) {
    __shared__ float xs[HD];
    int b = blockIdx.x, j = threadIdx.x;
    xs[j] = x[b * HD + j];
    __syncthreads();
    float acc = 0.f;
    for (int k = 0; k < HD; ++k) acc += xs[k] * W[(size_t)k * HD + j];
    out[b * HD + j] = acc + bias[j];
}

// ---------------------------------------------------------------------------
// Kernel: leaf log-softmax for the root row.  grid = B
// ---------------------------------------------------------------------------
__global__ __launch_bounds__(1024)
void k_leaf0(const float* __restrict__ Hall, const float* __restrict__ Wl,
             const float* __restrict__ bl, float* __restrict__ Lv) {
    __shared__ float red[2 * HD];
    int b = blockIdx.x, j = threadIdx.x;
    float h = Hall[b * HD + j];
    red[j]      = h * Wl[j * 2 + 0];
    red[HD + j] = h * Wl[j * 2 + 1];
    __syncthreads();
    for (int s = HD / 2; s > 0; s >>= 1) {
        if (j < s) { red[j] += red[j + s]; red[HD + j] += red[HD + j + s]; }
        __syncthreads();
    }
    if (j == 0) {
        float l0 = red[0] + bl[0], l1 = red[HD] + bl[1];
        float m  = fmaxf(l0, l1);
        float lse = m + logf(expf(l0 - m) + expf(l1 - m));
        Lv[b * 2 + 0] = l0 - lse;
        Lv[b * 2 + 1] = l1 - lse;
    }
}

// ---------------------------------------------------------------------------
// Kernel: one tree level.  grid = #surviving children, block = 1024 threads.
// child c (parent p, side s): z = [h_p, g] @ W_cell[:, s*H : (s+1)*H] + b_cell
//   branch = tanh(LN(z));  ll = 0.5*parent_ll[1] + log_softmax(branch@W_leaf+b_leaf)
// ---------------------------------------------------------------------------
__global__ __launch_bounds__(1024)
void k_cell_level(float* __restrict__ Hall, const float* __restrict__ g,
                  const float* __restrict__ Wc, const float* __restrict__ bc,
                  const float* __restrict__ lng, const float* __restrict__ lnb,
                  const float* __restrict__ Wl, const float* __restrict__ bl,
                  float* __restrict__ Lv,
                  const int* __restrict__ childP, const int* __restrict__ childS,
                  int poff, int coff) {
    __shared__ float smem[16 * HD];            // 64 KB: inp staging, then reductions
    int c = blockIdx.x;
    int p = childP[c], side = childS[c];
    int j = threadIdx.x;                       // 0..1023, one z-column each

    const float* hp = Hall + (size_t)(poff + p) * (BB * HD);
    // inp[b][k] : k<H -> parent hidden, else global transform g
    for (int idx = j; idx < BB * 2 * HD; idx += 1024) {
        int b = idx >> 11, k = idx & (2 * HD - 1);
        smem[idx] = (k < HD) ? hp[b * HD + k] : g[b * HD + (k - HD)];
    }
    __syncthreads();

    float acc[BB];
    #pragma unroll
    for (int b = 0; b < BB; ++b) acc[b] = 0.f;
    const float* wcol = Wc + (size_t)side * HD + j;        // column side*H + j
    for (int k = 0; k < 2 * HD; ++k) {
        float w = wcol[(size_t)k * (2 * HD)];
        #pragma unroll
        for (int b = 0; b < BB; ++b) acc[b] += smem[b * (2 * HD) + k] * w;
    }
    float bcv = bc[side * HD + j];
    #pragma unroll
    for (int b = 0; b < BB; ++b) acc[b] += bcv;
    __syncthreads();                                        // smem free -> reductions

    // LayerNorm stats: 16 simultaneous tree-reductions (sum, sumsq per b)
    #pragma unroll
    for (int b = 0; b < BB; ++b) {
        smem[b * HD + j]        = acc[b];
        smem[(BB + b) * HD + j] = acc[b] * acc[b];
    }
    __syncthreads();
    for (int s = HD / 2; s > 0; s >>= 1) {
        if (j < s) {
            #pragma unroll
            for (int q = 0; q < 16; ++q) smem[q * HD + j] += smem[q * HD + j + s];
        }
        __syncthreads();
    }
    float br[BB];
    float gj = lng[j], bj = lnb[j];
    #pragma unroll
    for (int b = 0; b < BB; ++b) {
        float mu  = smem[b * HD] * (1.f / HD);
        float var = smem[(BB + b) * HD] * (1.f / HD) - mu * mu;
        float rs  = rsqrtf(var + LN_EPS);
        br[b] = tanhf((acc[b] - mu) * rs * gj + bj);
    }
    __syncthreads();

    // write child hidden state
    float* ho = Hall + (size_t)(coff + c) * (BB * HD);
    #pragma unroll
    for (int b = 0; b < BB; ++b) ho[b * HD + j] = br[b];

    // leaf logits: two dots per b (16 simultaneous reductions)
    float w0 = Wl[j * 2 + 0], w1 = Wl[j * 2 + 1];
    #pragma unroll
    for (int b = 0; b < BB; ++b) {
        smem[b * HD + j]        = br[b] * w0;
        smem[(BB + b) * HD + j] = br[b] * w1;
    }
    __syncthreads();
    for (int s = HD / 2; s > 0; s >>= 1) {
        if (j < s) {
            #pragma unroll
            for (int q = 0; q < 16; ++q) smem[q * HD + j] += smem[q * HD + j + s];
        }
        __syncthreads();
    }
    if (j < BB) {
        int b = j;
        float l0 = smem[b * HD] + bl[0];
        float l1 = smem[(BB + b) * HD] + bl[1];
        float m  = fmaxf(l0, l1);
        float lse = m + logf(expf(l0 - m) + expf(l1 - m));
        float pl  = 0.5f * Lv[(size_t)(poff + p) * (BB * 2) + b * 2 + 1];
        Lv[(size_t)(coff + c) * (BB * 2) + b * 2 + 0] = pl + (l0 - lse);
        Lv[(size_t)(coff + c) * (BB * 2) + b * 2 + 1] = pl + (l1 - lse);
    }
}

// ---------------------------------------------------------------------------
// Pack A (hiddens, fp32 [M,1024]) into WMMA A-fragment order, bf16.
// Lane L (m = mtile*16 + L%16, khalf = L/16) element e:
//   k = ktile*32 + (e<8 ? khalf*8+e : 16 + khalf*8 + (e-8))      (ISA 16-bit A 16x32)
// grid = (MTp, 32), block = 32.  Rows >= Mrows are zero-filled.
// ---------------------------------------------------------------------------
__global__ __launch_bounds__(32)
void k_pack_a(const float* __restrict__ Hall, __bf16* __restrict__ Apack, int Mrows) {
    int mt = blockIdx.x, kt = blockIdx.y, lane = threadIdx.x;
    int m = mt * 16 + (lane & 15);
    int khalf = lane >> 4;
    v16bf vv;
    #pragma unroll
    for (int e = 0; e < 16; ++e) {
        int k = kt * 32 + ((e < 8) ? (khalf * 8 + e) : (16 + khalf * 8 + (e - 8)));
        float f = (m < Mrows) ? Hall[(size_t)m * HD + k] : 0.f;
        vv[e] = f2bf(f);
    }
    size_t off = (((size_t)mt * 32 + kt) * 32 + lane) * 16;
    *(v16bf*)(Apack + off) = vv;
}

// ---------------------------------------------------------------------------
// Pack B (W_out fp32 [1024,32000]) into WMMA B-fragment order, bf16.
// Lane L (n = ntile*16 + L%16): element e -> k = ktile*32 + (e<8 ? (L/16)*8+e : 16 + (L/16)*8 + (e-8))
// (dense analogue of the documented 64x16 16-bit B layout)
// grid = (2000, 32), block = 32
// ---------------------------------------------------------------------------
__global__ __launch_bounds__(32)
void k_pack_b(const float* __restrict__ Wout, __bf16* __restrict__ Bpack) {
    int nt = blockIdx.x, kt = blockIdx.y, lane = threadIdx.x;
    int n = nt * 16 + (lane & 15);
    int kbase = kt * 32 + (lane >> 4) * 8;
    v16bf vv;
    #pragma unroll
    for (int e = 0; e < 16; ++e)
        vv[e] = f2bf(Wout[(size_t)(kbase + ((e < 8) ? e : (8 + e))) * VV + n]);
    size_t off = (((size_t)nt * 32 + kt) * 32 + lane) * 16;
    *(v16bf*)(Bpack + off) = vv;
}

// ---------------------------------------------------------------------------
// WMMA GEMM:  logits[M, 32000] = A[M,1024] x B[1024,32000] + b_out
// Register tile: 4 m-tiles x 4 n-tiles per wave (16 accumulators).
// Per k-step: 8 A-frag + 8 B-frag b128 loads feed 16 WMMAs (1 load / WMMA,
// 32 FLOP/byte from L2).  Block = 4 waves, all sharing the same 4 m-tiles
// (A frags hit in L0) but different n-groups.
// grid = (MTp/4, 32000/(16*4*4) = 125)
// ---------------------------------------------------------------------------
__global__ __launch_bounds__(128)
void k_wmma(const __bf16* __restrict__ A, const __bf16* __restrict__ B,
            const float* __restrict__ bias, float* __restrict__ out,
            int Mrows) {
    int wave = threadIdx.x >> 5;
    int lane = threadIdx.x & 31;
    int mtg  = blockIdx.x;                     // group of 4 m-tiles
    int ng   = blockIdx.y * 4 + wave;          // group of 4 n-tiles (64 cols)

    const v16bf* Ap = (const v16bf*)A;         // fragment index: (mt*32+kt)*32+lane
    const v16bf* Bp = (const v16bf*)B;         // fragment index: (nt*32+kt)*32+lane

    v8f acc[4][4];
    #pragma unroll
    for (int i = 0; i < 4; ++i)
        #pragma unroll
        for (int j = 0; j < 4; ++j)
            acc[i][j] = (v8f){0,0,0,0,0,0,0,0};

    size_t abase[4], bbase[4];
    #pragma unroll
    for (int i = 0; i < 4; ++i)
        abase[i] = ((size_t)(mtg * 4 + i) * 32) * 32 + lane;
    #pragma unroll
    for (int j = 0; j < 4; ++j)
        bbase[j] = ((size_t)(ng * 4 + j) * 32) * 32 + lane;

    for (int kt = 0; kt < 32; ++kt) {
        v16bf a[4], b[4];
        #pragma unroll
        for (int i = 0; i < 4; ++i) a[i] = Ap[abase[i] + (size_t)kt * 32];
        #pragma unroll
        for (int j = 0; j < 4; ++j) b[j] = Bp[bbase[j] + (size_t)kt * 32];
        #pragma unroll
        for (int i = 0; i < 4; ++i)
            #pragma unroll
            for (int j = 0; j < 4; ++j)
                acc[i][j] = __builtin_amdgcn_wmma_f32_16x16x32_bf16(
                    false, a[i], false, b[j], (short)0, acc[i][j], false, false);
    }

    // D layout: VGPR r -> m = r + 8*(lane/16); n = lane%16
    int nb    = ng * 64 + (lane & 15);
    int khalf = lane >> 4;
    float bi[4];
    #pragma unroll
    for (int j = 0; j < 4; ++j) bi[j] = bias[nb + j * 16];
    #pragma unroll
    for (int i = 0; i < 4; ++i) {
        int mbase = (mtg * 4 + i) * 16 + 8 * khalf;
        #pragma unroll
        for (int r = 0; r < 8; ++r) {
            int m = mbase + r;
            if (m < Mrows) {
                float* row = out + (size_t)m * VV;
                #pragma unroll
                for (int j = 0; j < 4; ++j)
                    row[nb + j * 16] = acc[i][j][r] + bi[j];
            }
        }
    }
}

// ---------------------------------------------------------------------------
// Row-wise log-softmax over V=32000, in place, float4 (b128) accesses.
// grid = Mrows, block = 512
// ---------------------------------------------------------------------------
__global__ __launch_bounds__(512)
void k_lsm(float* __restrict__ out, int ncols) {
    __shared__ float red[512];
    int t = threadIdx.x;
    float4* p4 = (float4*)(out + (size_t)blockIdx.x * ncols);
    int n4 = ncols >> 2;

    float mx = -3.4e38f;
    for (int c = t; c < n4; c += 512) {
        float4 v = p4[c];
        mx = fmaxf(mx, fmaxf(fmaxf(v.x, v.y), fmaxf(v.z, v.w)));
    }
    red[t] = mx; __syncthreads();
    for (int s = 256; s > 0; s >>= 1) {
        if (t < s) red[t] = fmaxf(red[t], red[t + s]);
        __syncthreads();
    }
    mx = red[0]; __syncthreads();

    float sum = 0.f;
    for (int c = t; c < n4; c += 512) {
        float4 v = p4[c];
        sum += expf(v.x - mx) + expf(v.y - mx) + expf(v.z - mx) + expf(v.w - mx);
    }
    red[t] = sum; __syncthreads();
    for (int s = 256; s > 0; s >>= 1) {
        if (t < s) red[t] += red[t + s];
        __syncthreads();
    }
    float lse = mx + logf(red[0]);

    for (int c = t; c < n4; c += 512) {
        float4 v = p4[c];
        v.x -= lse; v.y -= lse; v.z -= lse; v.w -= lse;
        p4[c] = v;
    }
}

// ---------------------------------------------------------------------------
// kernel_launch
// ---------------------------------------------------------------------------
extern "C" void kernel_launch(void* const* d_in, const int* in_sizes, int n_in,
                              void* d_out, int out_size, void* d_ws, size_t ws_size,
                              hipStream_t stream) {
    (void)in_sizes; (void)n_in; (void)out_size; (void)ws_size;
    const float* x      = (const float*)d_in[0];
    const float* W_in   = (const float*)d_in[1];
    const float* b_in   = (const float*)d_in[2];
    const float* W_g    = (const float*)d_in[3];
    const float* b_g    = (const float*)d_in[4];
    const float* W_cell = (const float*)d_in[5];
    const float* b_cell = (const float*)d_in[6];
    const float* ln_g   = (const float*)d_in[7];
    const float* ln_b   = (const float*)d_in[8];
    const float* W_leaf = (const float*)d_in[9];
    const float* b_leaf = (const float*)d_in[10];
    const float* W_out  = (const float*)d_in[11];
    const float* b_out  = (const float*)d_in[12];
    // d_in[13] = depth (static = 14; structure precomputed at load)

    const omd::Structure& S = omd::g_S;
    const int T     = S.T;                       // 351
    const int Mrows = T * BB;                    // 2808
    const int MT    = (Mrows + 15) / 16;         // 176
    const int MTp   = ((MT + 3) / 4) * 4;        // padded to multiple of 4 m-tiles
    const int NT    = VV / 16;                   // 2000

    // workspace layout
    char* ws = (char*)d_ws;
    size_t off = 0;
    float* gbuf = (float*)(ws + off);  off = omd::alignUp(off + (size_t)BB * HD * 4, 256);
    float* Hall = (float*)(ws + off);  off = omd::alignUp(off + (size_t)T * BB * HD * 4, 256);
    __bf16* Apack = (__bf16*)(ws + off); off = omd::alignUp(off + (size_t)MTp * 32 * 32 * 16 * 2, 256);
    __bf16* Bpack = (__bf16*)(ws + off); off = omd::alignUp(off + (size_t)NT * 32 * 32 * 16 * 2, 256);
    int* dChildP = (int*)(ws + off);   off = omd::alignUp(off + S.childP.size() * 4, 256);
    int* dChildS = (int*)(ws + off);   off = omd::alignUp(off + S.childS.size() * 4, 256);

    hipMemcpyAsync(dChildP, S.childP.data(), S.childP.size() * 4, hipMemcpyHostToDevice, stream);
    hipMemcpyAsync(dChildS, S.childS.data(), S.childS.size() * 4, hipMemcpyHostToDevice, stream);

    float* out = (float*)d_out;
    float* Lv  = out + (size_t)T * BB * VV;      // leaves region [T][B][2]

    // static index outputs (bit-stored int32)
    size_t o = (size_t)T * BB * VV + (size_t)T * BB * 2;
    hipMemcpyAsync(out + o, S.Bl0.data(),  S.Bl0.size()  * 4, hipMemcpyHostToDevice, stream); o += S.Bl0.size();
    hipMemcpyAsync(out + o, S.Br0.data(),  S.Br0.size()  * 4, hipMemcpyHostToDevice, stream); o += S.Br0.size();
    hipMemcpyAsync(out + o, S.Sout.data(), S.Sout.size() * 4, hipMemcpyHostToDevice, stream); o += S.Sout.size();
    hipMemcpyAsync(out + o, S.Sin.data(),  S.Sin.size()  * 4, hipMemcpyHostToDevice, stream);

    // pack W_out early (independent of tree)
    k_pack_b<<<dim3(NT, 32), 32, 0, stream>>>(W_out, Bpack);

    // h0 = x@W_in + b_in (row 0 of Hall);  g = x@W_g + b_g
    k_in<<<BB, HD, 0, stream>>>(x, W_in, b_in, Hall);
    k_in<<<BB, HD, 0, stream>>>(x, W_g, b_g, gbuf);
    k_leaf0<<<BB, HD, 0, stream>>>(Hall, W_leaf, b_leaf, Lv);

    // 13 sequential tree levels
    for (size_t l = 0; l + 1 < S.levelK.size(); ++l) {
        int Kc = S.levelK[l + 1];
        if (Kc <= 0) continue;
        k_cell_level<<<Kc, HD, 0, stream>>>(
            Hall, gbuf, W_cell, b_cell, ln_g, ln_b, W_leaf, b_leaf, Lv,
            dChildP + S.childBase[l], dChildS + S.childBase[l],
            S.levelOff[l], S.levelOff[l + 1]);
    }

    // big output projection: pack A (zero-padded), WMMA GEMM, log-softmax
    k_pack_a<<<dim3(MTp, 32), 32, 0, stream>>>(Hall, Apack, Mrows);
    k_wmma<<<dim3(MTp / 4, VV / 256), 128, 0, stream>>>(Apack, Bpack, b_out, out, Mrows);
    k_lsm<<<Mrows, 512, 0, stream>>>(out, VV);
}
